// SDFNetwork_48653389529342
// MI455X (gfx1250) — matrix operand order repaired and checked by reading, not compiled, not run on hardware
//
#include <hip/hip_runtime.h>

typedef __attribute__((ext_vector_type(16))) _Float16 v16h;
typedef __attribute__((ext_vector_type(8)))  float    v8f;

#define LVLS 16
#define TSIZE (1u << 19)           // 2^19 entries per level
#define PPB 256                    // points per block (8 waves x 2 tiles x 16)

// A fragment (16x32 f16) from row-major LDS [rows][ld], rows m0..m0+15, K chunk kc.
__device__ __forceinline__ v16h ld_afrag(const _Float16* base, int ld, int m0, int kc, int lane) {
    const int m  = m0 + (lane & 15);
    const int kb = kc + ((lane >> 4) << 4);         // 0 or 16 within chunk
    const _Float16* rp = base + m * ld + kb;
    v16h a;
    ((float4*)&a)[0] = *(const float4*)(rp);        // halves K = kb .. kb+7
    ((float4*)&a)[1] = *(const float4*)(rp + 8);    // halves K = kb+8 .. kb+15
    return a;
}

// B fragment from pre-swizzled LDS: frag-major / lane-major / element-contiguous.
__device__ __forceinline__ v16h ld_bfrag_swz(const _Float16* sBF, int fid, int lane) {
    const _Float16* p = sBF + fid * 512 + lane * 16;
    v16h b;
    ((float4*)&b)[0] = *(const float4*)(p);
    ((float4*)&b)[1] = *(const float4*)(p + 8);
    return b;
}

__global__ __launch_bounds__(256) void sdf_hashgrid_mlp(
    const float* __restrict__ x,       // [N,3] in [-1,1]
    const float* __restrict__ tables,  // [16][2^19][2]
    const float* __restrict__ W1,      // [32,64]
    const float* __restrict__ W2,      // [64,64]
    const float* __restrict__ W3,      // [64,1]
    float* __restrict__ out, int N)
{
    __shared__ __align__(16) _Float16 sEnc[PPB * 32];   // 16 KB encodings
    __shared__ __align__(16) _Float16 sBF[12 * 512];    // 12 KB: 12 B-frags in WMMA layout
    __shared__ __align__(16) float    sW3[64];
    __shared__ __align__(16) _Float16 sH[8][16 * 64];   // 16 KB per-wave h1 buffer

    const int tid = threadIdx.x;

    // ---- stage weights into LDS, pre-swizzled into B-fragment order ----
    // B order used here (contiguous k, as ld_afrag): lane = (n&15) | 16*(kk>=16), element j = kk&15.
    for (int i = tid; i < 32 * 64; i += 256) {          // W1: frags 0..3 (k0=0, nc=n>>4)
        const int k = i >> 6, n = i & 63;
        const int fid    = n >> 4;
        const int lane16 = (n & 15) | ((k >> 4) << 4);
        sBF[fid * 512 + lane16 * 16 + (k & 15)] = (_Float16)W1[i];
    }
    for (int i = tid; i < 64 * 64; i += 256) {          // W2: frags 4..11 (kc=k>>5, nc=n>>4)
        const int k = i >> 6, n = i & 63;
        const int kk = k & 31;
        const int fid    = 4 + ((k >> 5) << 2) + (n >> 4);
        const int lane16 = (n & 15) | ((kk >> 4) << 4);
        sBF[fid * 512 + lane16 * 16 + (kk & 15)] = (_Float16)W2[i];
    }
    if (tid < 64) sW3[tid] = W3[tid];

    // ---- phase 1: multiresolution hash encoding, one point per thread ----
    const int gp = blockIdx.x * PPB + tid;
    {
        float xr = 0.f, yr = 0.f, zr = 0.f;
        if (gp < N) { xr = x[gp * 3 + 0]; yr = x[gp * 3 + 1]; zr = x[gp * 3 + 2]; }
        const float x0 = (xr + 1.f) * 0.5f;
        const float y0 = (yr + 1.f) * 0.5f;
        const float z0 = (zr + 1.f) * 0.5f;

        float g = 16.0f;                              // BASE * S^l
#pragma unroll 1
        for (int l = 0; l < LVLS; ++l) {
            const float scale = g - 1.0f;
            g *= 1.3819f;
            const int res = (int)ceilf(scale) + 1;
            const bool dense = (res <= 80);           // res^3 <= 2^19  <=>  res <= 80

            const float px = fmaf(x0, scale, 0.5f);
            const float py = fmaf(y0, scale, 0.5f);
            const float pz = fmaf(z0, scale, 0.5f);
            const float fx = floorf(px), fy = floorf(py), fz = floorf(pz);
            const float wx = px - fx, wy = py - fy, wz = pz - fz;
            const int cx0 = (int)fx, cy0 = (int)fy, cz0 = (int)fz;

            const float2* tbl = (const float2*)tables + (size_t)l * TSIZE;
            float a0 = 0.f, a1 = 0.f;
#pragma unroll
            for (int c = 0; c < 8; ++c) {
                const int dx = c & 1, dy = (c >> 1) & 1, dz = c >> 2;
                const int cx = min(max(cx0 + dx, 0), res - 1);
                const int cy = min(max(cy0 + dy, 0), res - 1);
                const int cz = min(max(cz0 + dz, 0), res - 1);
                int idx;
                if (dense) {
                    idx = cx + res * (cy + res * cz);
                } else {
                    const unsigned h = (unsigned)cx
                                     ^ ((unsigned)cy * 2654435761u)
                                     ^ ((unsigned)cz * 805459861u);
                    idx = (int)(h & (TSIZE - 1u));
                }
                const float wgt = (dx ? wx : 1.f - wx) * (dy ? wy : 1.f - wy)
                                * (dz ? wz : 1.f - wz);
                const float2 t = tbl[idx];            // L2-resident gather (64MB << 192MB)
                a0 = fmaf(t.x, wgt, a0);
                a1 = fmaf(t.y, wgt, a1);
            }
            sEnc[tid * 32 + 2 * l]     = (_Float16)a0;
            sEnc[tid * 32 + 2 * l + 1] = (_Float16)a1;
        }
    }
    __syncthreads();

    // ---- phase 2: WMMA MLP, one 16-point tile at a time per wave ----
    const int lane = tid & 31;
    const int w    = tid >> 5;
    _Float16* hbuf = sH[w];
    const int mrow = (lane >> 4) << 3;   // C/D: m = r + 8*(lane>=16)
    const int ncol = lane & 15;

    // hoisted B fragments (two ds_load_b128 each, conflict-free)
    v16h b1[4], b2a[4], b2b[4];
#pragma unroll
    for (int nc = 0; nc < 4; ++nc) {
        b1[nc]  = ld_bfrag_swz(sBF, nc,     lane);
        b2a[nc] = ld_bfrag_swz(sBF, 4 + nc, lane);
        b2b[nc] = ld_bfrag_swz(sBF, 8 + nc, lane);
    }
    // per-lane W3 values for this lane's 4 output columns
    float w3v[4];
#pragma unroll
    for (int nc = 0; nc < 4; ++nc) w3v[nc] = sW3[nc * 16 + ncol];

#pragma unroll 1
    for (int t = 0; t < 2; ++t) {
        const int m0 = w * 32 + t * 16;               // block-local row base

        // layer 1: enc(16x32) @ W1(32x64) -> relu -> hbuf(16x64)
        const v16h a1 = ld_afrag(sEnc, 32, m0, 0, lane);
#pragma unroll
        for (int nc = 0; nc < 4; ++nc) {
            v8f c = {};
            c = __builtin_amdgcn_wmma_f32_16x16x32_f16(false, a1, false, b1[nc],
                                                       (short)0, c, false, false);
#pragma unroll
            for (int r = 0; r < 8; ++r)
                hbuf[(r + mrow) * 64 + nc * 16 + ncol] = (_Float16)fmaxf(c[r], 0.f);
        }

        // layer 2: h1(16x64) @ W2(64x64), accumulate over two K chunks
        const v16h a2lo = ld_afrag(hbuf, 64, 0, 0,  lane);
        const v16h a2hi = ld_afrag(hbuf, 64, 0, 32, lane);
        v8f c2[4];
#pragma unroll
        for (int nc = 0; nc < 4; ++nc) {
            v8f c = {};
            c = __builtin_amdgcn_wmma_f32_16x16x32_f16(false, a2lo, false, b2a[nc],
                                                       (short)0, c, false, false);
            c = __builtin_amdgcn_wmma_f32_16x16x32_f16(false, a2hi, false, b2b[nc],
                                                       (short)0, c, false, false);
            c2[nc] = c;
        }

        // layer 3: relu(h2) @ W3 directly from C-layout registers.
        // Lane holds columns {nc*16+ncol} of rows mrow..mrow+7 -> per-lane partial dot,
        // then butterfly-reduce across the 16 lanes of each half-wave.
        float p[8];
#pragma unroll
        for (int r = 0; r < 8; ++r) {
            float s = 0.f;
#pragma unroll
            for (int nc = 0; nc < 4; ++nc)
                s = fmaf(fmaxf(c2[nc][r], 0.f), w3v[nc], s);
            p[r] = s;
        }
#pragma unroll
        for (int msk = 1; msk < 16; msk <<= 1) {
#pragma unroll
            for (int r = 0; r < 8; ++r)
                p[r] += __shfl_xor(p[r], msk, 32);
        }
        if ((lane & 15) == 0) {
            const int gbase = blockIdx.x * PPB + m0 + mrow;
#pragma unroll
            for (int r = 0; r < 8; ++r)
                if (gbase + r < N) out[gbase + r] = p[r];
        }
    }
}

extern "C" void kernel_launch(void* const* d_in, const int* in_sizes, int n_in,
                              void* d_out, int out_size, void* d_ws, size_t ws_size,
                              hipStream_t stream) {
    const float* x      = (const float*)d_in[0];
    const float* tables = (const float*)d_in[1];
    const float* W1     = (const float*)d_in[2];
    const float* W2     = (const float*)d_in[3];
    const float* W3     = (const float*)d_in[4];
    float* out = (float*)d_out;
    const int N = in_sizes[0] / 3;
    const int blocks = (N + PPB - 1) / PPB;
    sdf_hashgrid_mlp<<<blocks, PPB, 0, stream>>>(x, tables, W1, W2, W3, out, N);
}
